// ET_49160195670123
// MI455X (gfx1250) — matrix operand order repaired and checked by reading, not compiled, not run on hardware
//
#include <hip/hip_runtime.h>
#include <hip/hip_bf16.h>

#define BDIM 8
#define LDIM 64
#define DDIM 768
#define FDIM 50
#define PDIM 49
#define ADIM 49
#define NPTS 20000
#define CELLS 36
#define OUTROWS (FDIM + CELLS)
#define LN_EPS 1e-12f

typedef __attribute__((ext_vector_type(16))) __bf16 v16bf;
typedef __attribute__((ext_vector_type(8)))  float  v8f;

// ---- order-preserving float <-> uint encode for atomicMax on floats ----
__device__ __forceinline__ unsigned fenc(float f) {
  unsigned u = __float_as_uint(f);
  return (u & 0x80000000u) ? ~u : (u | 0x80000000u);
}
__device__ __forceinline__ float fdec(unsigned u) {
  return (u & 0x80000000u) ? __uint_as_float(u & 0x7FFFFFFFu)
                           : __uint_as_float(~u);
}
#define ENC_NEG_INF 0x007FFFFFu   // fenc(-inf)

// ---- WMMA bf16 fragment loaders ----
// A (16x32, MxK), fp32 source converted in-flight:
// lane&15 -> M; elems 0..15 -> K = hi*16 + 0..15.
__device__ __forceinline__ v16bf load_frag_A(const float* __restrict__ row, int hi) {
  v16bf a;
  const float* p0 = row + hi * 16;
  const float* p1 = row + hi * 16 + 8;
#pragma unroll
  for (int i = 0; i < 8; ++i) {
    a[i]     = (__bf16)p0[i];
    a[8 + i] = (__bf16)p1[i];
  }
  return a;
}
// B (32x16, KxN), pre-converted bf16 source: lane&15 -> N column;
// lanes 0-15 hold K=0..15, lanes 16-31 hold K=16..31 (16 contiguous bf16,
// 32-byte aligned -> two global_load_b128, zero converts).
__device__ __forceinline__ v16bf load_frag_Bbf(const __bf16* __restrict__ p, int hi) {
  return *(const v16bf*)(p + hi * 16);
}

// =====================================================================
// init: den=0, acc_g=0, m_enc = enc(-inf)
// =====================================================================
__global__ void init_kernel(unsigned* __restrict__ m_enc,
                            float* __restrict__ den,
                            float* __restrict__ acc_g) {
  int i = blockIdx.x * blockDim.x + threadIdx.x;
  if (i < BDIM * CELLS) { m_enc[i] = ENC_NEG_INF; den[i] = 0.f; }
  if (i < BDIM * CELLS * DDIM) acc_g[i] = 0.f;
}

// =====================================================================
// wconv: W_grid fp32 -> bf16 (one-time)
// =====================================================================
__global__ void wconv_kernel(const float* __restrict__ W_grid,
                             __bf16* __restrict__ wgrid_bf) {
  int i = blockIdx.x * blockDim.x + threadIdx.x;
  if (i < DDIM * DDIM) wgrid_bf[i] = (__bf16)W_grid[i];
}

// =====================================================================
// text = lang @ W_text.T + b_text    [B,L,D] -> bf16 (tiny)
// =====================================================================
__global__ void text_kernel(const float* __restrict__ lang,
                            const float* __restrict__ W_text,
                            const float* __restrict__ b_text,
                            __bf16* __restrict__ text_bf) {
  const int l = blockIdx.x, b = blockIdx.y;
  const float* lrow = lang + ((size_t)b * LDIM + l) * DDIM;
  for (int j = threadIdx.x; j < DDIM; j += blockDim.x) {
    const float* wr = W_text + (size_t)j * DDIM;
    float s = b_text[j];
    for (int k = 0; k < DDIM; ++k) s += lrow[k] * wr[k];
    text_bf[((size_t)b * LDIM + l) * DDIM + j] = (__bf16)s;
  }
}

// =====================================================================
// frames branch: SoftDotAttention + fc2  -> out rows [0, F)
// =====================================================================
__global__ void frames_kernel(const float* __restrict__ lang_cls,
                              const float* __restrict__ frames,
                              const float* __restrict__ W_in,
                              const float* __restrict__ W_out,
                              const float* __restrict__ W_fc2,
                              const float* __restrict__ b_fc2,
                              float* __restrict__ out) {
  const int f = blockIdx.x, b = blockIdx.y;
  const int tid = threadIdx.x;
  __shared__ float lc[ADIM], tgt[ADIM], at[PDIM], wctx[ADIM], af[ADIM];

  if (tid < ADIM) lc[tid] = lang_cls[b * ADIM + tid];
  __syncthreads();
  if (tid < ADIM) {
    float s = 0.f;
    for (int k = 0; k < ADIM; ++k) s += W_in[tid * ADIM + k] * lc[k];
    tgt[tid] = s;
  }
  __syncthreads();
  const float* fr = frames + (((size_t)b * FDIM + f) * PDIM) * ADIM;
  if (tid < PDIM) {
    float s = 0.f;
    for (int k = 0; k < ADIM; ++k) s += fr[tid * ADIM + k] * tgt[k];
    at[tid] = s;
  }
  __syncthreads();
  if (tid == 0) {                       // softmax over 49 (serial is cheap)
    float m = -3.0e38f;
    for (int p = 0; p < PDIM; ++p) m = fmaxf(m, at[p]);
    float s = 0.f;
    for (int p = 0; p < PDIM; ++p) { at[p] = __expf(at[p] - m); s += at[p]; }
    float inv = 1.f / s;
    for (int p = 0; p < PDIM; ++p) at[p] *= inv;
  }
  __syncthreads();
  if (tid < ADIM) {
    float s = 0.f;
    for (int p = 0; p < PDIM; ++p) s += at[p] * fr[p * ADIM + tid];
    wctx[tid] = s;
  }
  __syncthreads();
  if (tid < ADIM) {
    float s = 0.f;
    for (int k = 0; k < ADIM; ++k) s += wctx[k] * W_out[tid * (2 * ADIM) + k];
    for (int k = 0; k < ADIM; ++k) s += lc[k]   * W_out[tid * (2 * ADIM) + ADIM + k];
    af[tid] = tanhf(s);
  }
  __syncthreads();
  for (int j = tid; j < DDIM; j += blockDim.x) {
    float s = b_fc2[j];
    for (int k = 0; k < ADIM; ++k) s += af[k] * W_fc2[j * ADIM + k];
    out[((size_t)b * OUTROWS + f) * DDIM + j] = s;
  }
}

// =====================================================================
// scores: w[b,n] = max_l (grid_fts[b,n] . text[b,l]),  per-cell max
// WMMA GEMM M=20000, N=64, K=768 (bf16). 8 waves x 16 rows per block.
// =====================================================================
__global__ void scores_kernel(const float* __restrict__ grid_fts,
                              const __bf16* __restrict__ text_bf,
                              const int* __restrict__ grid_index,
                              float* __restrict__ w_out,
                              unsigned* __restrict__ m_enc) {
  const int b    = blockIdx.y;
  const int wave = threadIdx.x >> 5;
  const int lane = threadIdx.x & 31;
  const int hi   = lane >> 4;
  const int ln   = lane & 15;
  const int rowbase = blockIdx.x * 128 + wave * 16;
  if (rowbase + 16 > NPTS) return;                 // wave-uniform exit

  const float* grow = grid_fts + ((size_t)b * NPTS + rowbase + ln) * DDIM;
  const __bf16* tb  = text_bf + (size_t)b * LDIM * DDIM;
  const __bf16* t0  = tb + (size_t)(0 * 16 + ln) * DDIM;
  const __bf16* t1  = tb + (size_t)(1 * 16 + ln) * DDIM;
  const __bf16* t2  = tb + (size_t)(2 * 16 + ln) * DDIM;
  const __bf16* t3  = tb + (size_t)(3 * 16 + ln) * DDIM;

  v8f acc0 = {}, acc1 = {}, acc2 = {}, acc3 = {};
#pragma unroll 1
  for (int kk = 0; kk < DDIM; kk += 32) {
    __builtin_prefetch(grow + kk + 32, 0, 1);      // global_prefetch_b8
    v16bf a  = load_frag_A(grow + kk, hi);
    v16bf b0 = load_frag_Bbf(t0 + kk, hi);
    v16bf b1 = load_frag_Bbf(t1 + kk, hi);
    v16bf b2 = load_frag_Bbf(t2 + kk, hi);
    v16bf b3 = load_frag_Bbf(t3 + kk, hi);
    acc0 = __builtin_amdgcn_wmma_f32_16x16x32_bf16(false, a, false, b0, (short)0, acc0, false, false);
    acc1 = __builtin_amdgcn_wmma_f32_16x16x32_bf16(false, a, false, b1, (short)0, acc1, false, false);
    acc2 = __builtin_amdgcn_wmma_f32_16x16x32_bf16(false, a, false, b2, (short)0, acc2, false, false);
    acc3 = __builtin_amdgcn_wmma_f32_16x16x32_bf16(false, a, false, b3, (short)0, acc3, false, false);
  }

  // per-row max over 64 token columns: C layout lane=N, VGPR v -> M=v+8*hi
  float wm[8];
#pragma unroll
  for (int v = 0; v < 8; ++v) {
    float x = fmaxf(fmaxf(acc0[v], acc1[v]), fmaxf(acc2[v], acc3[v]));
    x = fmaxf(x, __shfl_xor(x, 1, 32));
    x = fmaxf(x, __shfl_xor(x, 2, 32));
    x = fmaxf(x, __shfl_xor(x, 4, 32));
    x = fmaxf(x, __shfl_xor(x, 8, 32));            // stays inside 16-lane half
    wm[v] = x;
  }
  if (ln == 0) {                                   // lanes 0 (rows +0..7) and 16 (rows +8..15)
#pragma unroll
    for (int v = 0; v < 8; ++v) {
      int row = rowbase + v + 8 * hi;
      w_out[(size_t)b * NPTS + row] = wm[v];
      int cell = grid_index[(size_t)b * NPTS + row];
      atomicMax(&m_enc[b * CELLS + cell], fenc(wm[v]));
    }
  }
}

// =====================================================================
// ew: e = exp(w - m[cell]); den[cell] += e
// =====================================================================
__global__ void ew_kernel(const float* __restrict__ wbuf,
                          const unsigned* __restrict__ m_enc,
                          const int* __restrict__ gidx,
                          float* __restrict__ ebuf,
                          float* __restrict__ den) {
  int i = blockIdx.x * blockDim.x + threadIdx.x;
  if (i >= BDIM * NPTS) return;
  int b = i / NPTS;
  int cell = gidx[i];
  float m = fdec(m_enc[b * CELLS + cell]);
  float e = __expf(wbuf[i] - m);
  ebuf[i] = e;
  atomicAdd(&den[b * CELLS + cell], e);
}

// =====================================================================
// gridmap: acc_g[b,c,:] += e[n] * (grid_fts[b,n] @ W_grid.T) for cell(n)=c
// WMMA GEMM M=20000, N=768, K=768 (bf16). Block = 256 rows, 8 waves x
// 6 column-tiles each; per-block 36x768 LDS accumulator (dynamic LDS).
// =====================================================================
__global__ void gridmap_kernel(const float* __restrict__ grid_fts,
                               const __bf16* __restrict__ wgrid_bf,
                               const int* __restrict__ grid_index,
                               const float* __restrict__ e_buf,
                               float* __restrict__ acc_g) {
  extern __shared__ float accLDS[];                // CELLS * DDIM floats
  const int b    = blockIdx.y;
  const int wave = threadIdx.x >> 5;
  const int lane = threadIdx.x & 31;
  const int hi   = lane >> 4;
  const int ln   = lane & 15;
  const int chunk = blockIdx.x * 256;
  const int nrows = min(256, NPTS - chunk);

  for (int i = threadIdx.x; i < CELLS * DDIM; i += blockDim.x) accLDS[i] = 0.f;
  __syncthreads();

  // per-lane B row pointers (constant across row groups)
  const __bf16* wg[6];
#pragma unroll
  for (int u = 0; u < 6; ++u)
    wg[u] = wgrid_bf + (size_t)((wave * 6 + u) * 16 + ln) * DDIM;

  for (int ri = 0; ri < nrows; ri += 16) {
    const int base = chunk + ri;
    const float* grow = grid_fts + ((size_t)b * NPTS + base + ln) * DDIM;

    v8f acc[6] = {};
#pragma unroll 1
    for (int kk = 0; kk < DDIM; kk += 32) {
      __builtin_prefetch(grow + kk + 32, 0, 1);
      v16bf a = load_frag_A(grow + kk, hi);
#pragma unroll
      for (int u = 0; u < 6; ++u) {
        v16bf bf = load_frag_Bbf(wg[u] + kk, hi);
        acc[u] = __builtin_amdgcn_wmma_f32_16x16x32_bf16(false, a, false, bf, (short)0, acc[u], false, false);
      }
    }

    float ev[8]; int cv[8];
#pragma unroll
    for (int v = 0; v < 8; ++v) {
      int row = base + v + 8 * hi;
      ev[v] = e_buf[(size_t)b * NPTS + row];
      cv[v] = grid_index[(size_t)b * NPTS + row];
    }
#pragma unroll
    for (int u = 0; u < 6; ++u) {
      int j = (wave * 6 + u) * 16 + ln;
#pragma unroll
      for (int v = 0; v < 8; ++v)
        atomicAdd(&accLDS[cv[v] * DDIM + j], acc[u][v] * ev[v]);
    }
  }
  __syncthreads();
  for (int i = threadIdx.x; i < CELLS * DDIM; i += blockDim.x)
    atomicAdd(&acc_g[(size_t)b * CELLS * DDIM + i], accLDS[i]);
}

// =====================================================================
// finalize: candidate Linear+LayerNorm, * lang[:,0,:], + grid_map
// out rows [F, F+C)
// =====================================================================
__global__ void finalize_kernel(const float* __restrict__ cand,
                                const float* __restrict__ lang,
                                const float* __restrict__ W_cand,
                                const float* __restrict__ b_cand,
                                const float* __restrict__ ln_g,
                                const float* __restrict__ ln_b,
                                const float* __restrict__ b_grid,
                                const float* __restrict__ acc_g,
                                const float* __restrict__ den,
                                float* __restrict__ out) {
  const int c = blockIdx.x, b = blockIdx.y;
  const int tid = threadIdx.x;
  __shared__ float red[256];
  __shared__ float s_mu, s_rstd;

  float x = cand[((size_t)b * CELLS + c) * 2 + 0];
  float y = cand[((size_t)b * CELLS + c) * 2 + 1];

  float vals[3];
  float s = 0.f;
#pragma unroll
  for (int i = 0; i < 3; ++i) {
    int j = tid + i * 256;
    float v = W_cand[j * 2] * x + W_cand[j * 2 + 1] * y + b_cand[j];
    vals[i] = v; s += v;
  }
  red[tid] = s; __syncthreads();
  for (int st = 128; st > 0; st >>= 1) {
    if (tid < st) red[tid] += red[tid + st];
    __syncthreads();
  }
  if (tid == 0) s_mu = red[0] / (float)DDIM;
  __syncthreads();
  float mu = s_mu;

  float s2 = 0.f;
#pragma unroll
  for (int i = 0; i < 3; ++i) { float d = vals[i] - mu; s2 += d * d; }
  red[tid] = s2; __syncthreads();
  for (int st = 128; st > 0; st >>= 1) {
    if (tid < st) red[tid] += red[tid + st];
    __syncthreads();
  }
  if (tid == 0) s_rstd = rsqrtf(red[0] / (float)DDIM + LN_EPS);
  __syncthreads();
  float rstd = s_rstd;

  float dv  = den[b * CELLS + c];
  float inv = (dv > 0.f) ? 1.f / fmaxf(dv, 1e-30f) : 0.f;
#pragma unroll
  for (int i = 0; i < 3; ++i) {
    int j = tid + i * 256;
    float nv = (vals[i] - mu) * rstd * ln_g[j] + ln_b[j];
    float gm = (dv > 0.f) ? acc_g[((size_t)b * CELLS + c) * DDIM + j] * inv + b_grid[j] : 0.f;
    out[((size_t)b * OUTROWS + FDIM + c) * DDIM + j] =
        nv * lang[(size_t)b * LDIM * DDIM + j] + gm;
  }
}

// =====================================================================
extern "C" void kernel_launch(void* const* d_in, const int* in_sizes, int n_in,
                              void* d_out, int out_size, void* d_ws, size_t ws_size,
                              hipStream_t stream) {
  const float* lang     = (const float*)d_in[0];
  const float* lang_cls = (const float*)d_in[1];
  const float* frames   = (const float*)d_in[2];
  const float* cand     = (const float*)d_in[3];
  const float* grid_fts = (const float*)d_in[4];
  const int*   gidx     = (const int*)d_in[5];
  const float* W_in     = (const float*)d_in[6];
  const float* W_out    = (const float*)d_in[7];
  const float* W_fc2    = (const float*)d_in[8];
  const float* b_fc2    = (const float*)d_in[9];
  const float* W_text   = (const float*)d_in[10];
  const float* b_text   = (const float*)d_in[11];
  const float* W_grid   = (const float*)d_in[12];
  const float* b_grid   = (const float*)d_in[13];
  const float* W_cand   = (const float*)d_in[14];
  const float* b_cand   = (const float*)d_in[15];
  const float* ln_g     = (const float*)d_in[16];
  const float* ln_b     = (const float*)d_in[17];
  float* out = (float*)d_out;

  // workspace layout: bf16 buffers first (32B-aligned), then fp32/uint
  char* wsb = (char*)d_ws;
  __bf16* text_bf  = (__bf16*)wsb;                                   // B*L*D bf16
  __bf16* wgrid_bf = (__bf16*)(wsb + (size_t)BDIM * LDIM * DDIM * 2);// D*D bf16
  float* wbuf = (float*)(wsb + (size_t)BDIM * LDIM * DDIM * 2
                             + (size_t)DDIM * DDIM * 2);             // B*N
  float* ebuf = wbuf + (size_t)BDIM * NPTS;                          // B*N
  unsigned* m_enc = (unsigned*)(ebuf + (size_t)BDIM * NPTS);         // B*C
  float* den   = (float*)(m_enc + BDIM * CELLS);                     // B*C
  float* acc_g = den + BDIM * CELLS;                                 // B*C*D

  init_kernel<<<(BDIM * CELLS * DDIM + 255) / 256, 256, 0, stream>>>(m_enc, den, acc_g);

  wconv_kernel<<<(DDIM * DDIM + 255) / 256, 256, 0, stream>>>(W_grid, wgrid_bf);

  text_kernel<<<dim3(LDIM, BDIM), 256, 0, stream>>>(lang, W_text, b_text, text_bf);

  frames_kernel<<<dim3(FDIM, BDIM), 256, 0, stream>>>(
      lang_cls, frames, W_in, W_out, W_fc2, b_fc2, out);

  scores_kernel<<<dim3((NPTS + 127) / 128, BDIM), 256, 0, stream>>>(
      grid_fts, text_bf, gidx, wbuf, m_enc);

  ew_kernel<<<(BDIM * NPTS + 255) / 256, 256, 0, stream>>>(
      wbuf, m_enc, gidx, ebuf, den);

  gridmap_kernel<<<dim3((NPTS + 255) / 256, BDIM), 256,
                   CELLS * DDIM * sizeof(float), stream>>>(
      grid_fts, wgrid_bf, gidx, ebuf, acc_g);

  finalize_kernel<<<dim3(CELLS, BDIM), 256, 0, stream>>>(
      cand, lang, W_cand, b_cand, ln_g, ln_b, b_grid, acc_g, den, out);
}
